// HybridAttention_16561393893624
// MI455X (gfx1250) — matrix operand order repaired and checked by reading, not compiled, not run on hardware
//
#include <hip/hip_runtime.h>
#include <stdint.h>

// ---------------------------------------------------------------------------
// MI455X (gfx1250) hybrid attention block.
// bf16 V_WMMA_F32_16X16X32_BF16 everywhere (8x K-depth of the f32 WMMA),
// f32 accumulation. Tile staging via Tensor Data Mover (tensor_load_to_lds,
// TENSORcnt) with double-buffered LDS when the builtin is available.
// ---------------------------------------------------------------------------

typedef unsigned short u16;
typedef __bf16 v16bf __attribute__((ext_vector_type(16)));
typedef float  v8f   __attribute__((ext_vector_type(8)));

union FragA { v16bf v; uint4 q[2]; };

__device__ __forceinline__ u16 f2bf(float f) {
  union { float f; unsigned u; } c; c.f = f;
  unsigned u = c.u;
  unsigned r = u + 0x7fffu + ((u >> 16) & 1u);   // round-to-nearest-even
  return (u16)(r >> 16);
}

constexpr int D   = 2048;
constexpr int Hh  = 16;
constexpr int HD  = 128;
constexpr int W   = 2048;
constexpr int T   = 2048;
constexpr int Bb  = 2;
constexpr long long Mrows = (long long)Bb * T;          // 4096
constexpr long long NX    = Mrows * D;                  // 8388608
constexpr long long NKV   = (long long)Bb * Hh * W * HD;// 8388608

// ---------------------------------------------------------------------------
// Tensor Data Mover: 2D bf16 tile -> LDS. D# built per CDNA5 ISA 8.3/8.4.
// This toolchain exposes the 6-arg builtin form:
//   (uint32x4 g0, int32x8 g1, int32x4 g2, int32x4 g3, int32x8 g4, i32 cpol)
// ---------------------------------------------------------------------------
#if defined(__has_builtin)
#if __has_builtin(__builtin_amdgcn_tensor_load_to_lds) && \
    __has_builtin(__builtin_amdgcn_s_wait_tensorcnt)
#define HAVE_TDM 1
#endif
#endif

#ifdef HAVE_TDM
typedef unsigned int u32x4 __attribute__((ext_vector_type(4)));
typedef int          i32x4 __attribute__((ext_vector_type(4)));
typedef int          i32x8 __attribute__((ext_vector_type(8)));

// Load a [tile_d1 rows x tile_d0 elems] bf16 tile (row stride stride0 elems)
// from global into LDS (rows packed contiguously). Issue once per workgroup.
__device__ __forceinline__ void tdm_load_tile_bf16(const u16* gsrc,
                                                   const u16* lds_dst,
                                                   int tile_d0, int tile_d1,
                                                   long long stride0) {
  unsigned long long ga = (unsigned long long)gsrc;
  unsigned lds = (unsigned)(unsigned long long)(const void*)lds_dst;
  const unsigned td0 = 0x40000000u, td1 = 0x40000000u;  // huge: never OOB-zeroed

  u32x4 g0;
  g0[0] = 1u;                                            // count=1, user mode
  g0[1] = lds;                                           // lds_addr (bytes)
  g0[2] = (unsigned)(ga & 0xffffffffu);                  // global_addr[31:0]
  g0[3] = (unsigned)((ga >> 32) & 0x01ffffffu) | (2u << 30); // ga[56:32]|type=2

  i32x8 g1;
  g1[0] = (int)(1u << 16);                               // mask=0, data_size=2B
  g1[1] = (int)((td0 & 0xffffu) << 16);                  // tensor_dim0 lo16
  g1[2] = (int)((td0 >> 16) | ((td1 & 0xffffu) << 16));  // td0 hi | td1 lo
  g1[3] = (int)((td1 >> 16) | ((unsigned)tile_d0 << 16));// td1 hi | tile_dim0
  g1[4] = (int)((unsigned)tile_d1);                      // tile_dim1, tile_dim2=0
  g1[5] = (int)((unsigned long long)stride0 & 0xffffffffu); // dim0 stride lo
  g1[6] = (int)(((unsigned long long)stride0 >> 32) & 0xffffu); // stride hi
  g1[7] = 0;                                             // dim1 stride (2D: unused)

  i32x4 g2 = {0, 0, 0, 0};                               // 2D: dims 2/3 unused
  i32x4 g3 = {0, 0, 0, 0};
  i32x8 g4 = {0, 0, 0, 0, 0, 0, 0, 0};                   // extra group (unused)
  __builtin_amdgcn_tensor_load_to_lds(g0, g1, g2, g3, g4, 0);
}
#endif

// ---------------------------------------------------------------------------
__global__ void k_init(unsigned* s) {
  if (threadIdx.x < 4) s[threadIdx.x] = 0u;
}

__global__ void k_absmax(const float* __restrict__ x, long long n,
                         unsigned* __restrict__ maxbits) {
  float m = 0.f;
  long long i = (long long)blockIdx.x * blockDim.x + threadIdx.x;
  long long stride = (long long)gridDim.x * blockDim.x;
  for (; i < n; i += stride) m = fmaxf(m, fabsf(x[i]));
  __shared__ float red[256];
  red[threadIdx.x] = m;
  __syncthreads();
  for (int s = 128; s > 0; s >>= 1) {
    if (threadIdx.x < s) red[threadIdx.x] = fmaxf(red[threadIdx.x], red[threadIdx.x + s]);
    __syncthreads();
  }
  if (threadIdx.x == 0) {
    union { float f; unsigned u; } c; c.f = red[0];
    atomicMax(maxbits, c.u);   // values >= 0 -> bit order == value order
  }
}

__global__ void k_quant_bf16(const float* __restrict__ x,
                             const unsigned* __restrict__ maxbits,
                             u16* __restrict__ out, long long n) {
  union { unsigned u; float f; } c; c.u = *maxbits;
  float scale = c.f / 127.0f + 1e-8f;
  long long i = (long long)blockIdx.x * blockDim.x + threadIdx.x;
  long long stride = (long long)gridDim.x * blockDim.x;
  for (; i < n; i += stride) {
    float q = rintf(x[i] / scale);
    q = fminf(fmaxf(q, -128.f), 127.f) * scale;
    out[i] = f2bf(q);
  }
}

__global__ void k_cvt_bf16(const float* __restrict__ x, u16* __restrict__ out,
                           long long n) {
  long long i = (long long)blockIdx.x * blockDim.x + threadIdx.x;
  long long stride = (long long)gridDim.x * blockDim.x;
  for (; i < n; i += stride) out[i] = f2bf(x[i]);
}

// ---------------------------------------------------------------------------
// Tiled bf16 WMMA GEMM: C[M x N] = A[M x K] * Bw[N x K]^T + bias.
// Block tile 128x128 (8 waves, 2x4), wave tile 64x32 (4x2 16x16 frags).
// TDM path: double-buffered LDS, one barrier/iteration, DMA overlaps WMMA.
// ---------------------------------------------------------------------------
__global__ __launch_bounds__(256)
void k_gemm_bf16(const u16* __restrict__ A, const u16* __restrict__ Bw,
                 const float* __restrict__ bias, int N, int K, int mode,
                 float* __restrict__ outC,
                 u16* __restrict__ qbf, float* __restrict__ kf32,
                 float* __restrict__ vf32, u16* __restrict__ kbf,
                 u16* __restrict__ vbf) {
#ifdef HAVE_TDM
  __shared__ __align__(16) u16 sA[2][128 * 32];
  __shared__ __align__(16) u16 sB[2][128 * 32];
#else
  __shared__ __align__(16) u16 sA[1][128 * 32];
  __shared__ __align__(16) u16 sB[1][128 * 32];
#endif

  const int tid  = threadIdx.x;
  const int lane = tid & 31;
  const int wave = tid >> 5;
  const int L15  = lane & 15;
  const int hi   = lane >> 4;
  const int wm   = wave >> 2;   // 0..1
  const int wn   = wave & 3;    // 0..3
  const long long rowBase = (long long)blockIdx.y * 128;
  const long long colBase = (long long)blockIdx.x * 128;

  const v8f zero = {0.f, 0.f, 0.f, 0.f, 0.f, 0.f, 0.f, 0.f};
  v8f acc[4][2];
#pragma unroll
  for (int i = 0; i < 4; ++i)
#pragma unroll
    for (int j = 0; j < 2; ++j) acc[i][j] = zero;

  auto compute_tile = [&](const u16* cA, const u16* cB) {
    FragA fb[2];
#pragma unroll
    for (int nf = 0; nf < 2; ++nf) {
      const u16* p = cB + (wn * 32 + nf * 16 + L15) * 32;
      fb[nf].q[0] = *reinterpret_cast<const uint4*>(p + hi * 8);
      fb[nf].q[1] = *reinterpret_cast<const uint4*>(p + 16 + hi * 8);
    }
#pragma unroll
    for (int mf = 0; mf < 4; ++mf) {
      FragA fa;
      const u16* ap = cA + (wm * 64 + mf * 16 + L15) * 32;
      fa.q[0] = *reinterpret_cast<const uint4*>(ap + hi * 8);
      fa.q[1] = *reinterpret_cast<const uint4*>(ap + 16 + hi * 8);
#pragma unroll
      for (int nf = 0; nf < 2; ++nf)
        acc[mf][nf] = __builtin_amdgcn_wmma_f32_16x16x32_bf16(
            false, fa.v, false, fb[nf].v, (short)0, acc[mf][nf], false, false);
    }
  };

#ifdef HAVE_TDM
  const int nIter = K / 32;
  if (tid < 32) {   // one wave issues the DMA descriptors
    tdm_load_tile_bf16(A + rowBase * K, sA[0], 32, 128, K);
    tdm_load_tile_bf16(Bw + colBase * K, sB[0], 32, 128, K);
  }
#pragma unroll 1
  for (int it = 0; it < nIter; ++it) {
    const int cur = it & 1;
    if (tid < 32) __builtin_amdgcn_s_wait_tensorcnt(0);
    __syncthreads();   // tile `it` ready; everyone done with tile `it-1`
    if (it + 1 < nIter && tid < 32) {
      const int k0 = (it + 1) * 32;
      tdm_load_tile_bf16(A + rowBase * K + k0, sA[cur ^ 1], 32, 128, K);
      tdm_load_tile_bf16(Bw + colBase * K + k0, sB[cur ^ 1], 32, 128, K);
    }
    compute_tile(sA[cur], sB[cur]);   // overlaps with TDM filling other buffer
  }
#else
  const int ldRow = tid >> 1;   // 0..127
  const int ldSeg = tid & 1;    // 2 x uint4 each
#pragma unroll 1
  for (int k0 = 0; k0 < K; k0 += 32) {
    const uint4* ga =
        reinterpret_cast<const uint4*>(A + (rowBase + ldRow) * K + k0) + ldSeg * 2;
    uint4* la = reinterpret_cast<uint4*>(sA[0] + ldRow * 32) + ldSeg * 2;
    la[0] = ga[0]; la[1] = ga[1];
    const uint4* gb =
        reinterpret_cast<const uint4*>(Bw + (colBase + ldRow) * K + k0) + ldSeg * 2;
    uint4* lb = reinterpret_cast<uint4*>(sB[0] + ldRow * 32) + ldSeg * 2;
    lb[0] = gb[0]; lb[1] = gb[1];
    __syncthreads();
    compute_tile(sA[0], sB[0]);
    __syncthreads();
  }
#endif

  // epilogue: C layout -> M = vgpr + 8*hi, N = lane&15   (fully unrolled:
  // constant vector indices, no v_movrels waterfall)
#pragma unroll
  for (int mf = 0; mf < 4; ++mf) {
#pragma unroll
    for (int nf = 0; nf < 2; ++nf) {
      long long col = colBase + wn * 32 + nf * 16 + L15;
      float bval = bias[col];
#pragma unroll
      for (int j = 0; j < 8; ++j) {
        long long row = rowBase + wm * 64 + mf * 16 + j + hi * 8;
        float val = acc[mf][nf][j] + bval;
        if (mode == 0) {
          outC[row * N + col] = val;
        } else {
          int which = (int)(col >> 11);
          int d = (int)(col & 2047);
          int h = d >> 7, hd = d & 127;
          int b = (int)(row >> 11), t = (int)(row & 2047);
          long long o = ((long long)(b * Hh + h) * W + t) * HD + hd;
          if (which == 0) {
            qbf[o] = f2bf(val);
          } else if (which == 1) {
            kf32[o] = val; kbf[o] = f2bf(val);
          } else {
            vf32[o] = val; vbf[o] = f2bf(val);
          }
        }
      }
    }
  }
}

// ---------------------------------------------------------------------------
// Flash attention with ALiBi bias, NO causal mask (matches reference).
// grid = (T/128, B*H). 8 waves x 16 query rows. Q frags in regs, K tile via
// TDM, V staged transposed, P transposed through per-wave LDS. Online softmax
// via 16-lane shuffle reductions.
// ---------------------------------------------------------------------------
__global__ __launch_bounds__(256)
void k_attn(const u16* __restrict__ qbf, const u16* __restrict__ kbf,
            const u16* __restrict__ vbf, const int* __restrict__ pos,
            float* __restrict__ outf) {
  __shared__ __align__(16) u16 sK[64 * 128];
  __shared__ __align__(16) u16 sVT[128 * 64];
  __shared__ __align__(16) u16 sP[8 * 16 * 64];

  const int tid  = threadIdx.x;
  const int lane = tid & 31;
  const int wave = tid >> 5;
  const int L15  = lane & 15;
  const int hi   = lane >> 4;
  const int bh   = blockIdx.y;        // 0..31
  const int h    = bh & 15;
  const int m0   = blockIdx.x * 128 + wave * 16;
  const long long qb = (long long)bh * T * HD;
  const long long kb = (long long)bh * W * HD;

  FragA aq[4];
  {
    const u16* qp = qbf + qb + (long long)(m0 + L15) * HD;
#pragma unroll
    for (int kk = 0; kk < 4; ++kk) {
      aq[kk].q[0] = *reinterpret_cast<const uint4*>(qp + kk * 32 + hi * 8);
      aq[kk].q[1] = *reinterpret_cast<const uint4*>(qp + kk * 32 + 16 + hi * 8);
    }
  }

  const float ascale = 0.08838834764831845f;           // 128^-0.5
  const float slope  = exp2f(-0.5f * (float)(h + 1));  // 2^(-8/H*(h+1))
  const float shift  = (float)(pos[0] + T - 1);        // new_pos - 1

  float mrow[8], lrow[8], alpha[8];
  const v8f zero = {0.f, 0.f, 0.f, 0.f, 0.f, 0.f, 0.f, 0.f};
  v8f oacc[8];
#pragma unroll
  for (int j = 0; j < 8; ++j) { mrow[j] = -3.0e38f; lrow[j] = 0.f; }
#pragma unroll
  for (int f = 0; f < 8; ++f) oacc[f] = zero;

  const int ldRow  = tid >> 2;  // 0..63 (key index in chunk)
  const int ldPart = tid & 3;   // 4 x uint4 each

#pragma unroll 1
  for (int w0 = 0; w0 < W; w0 += 64) {
#ifdef HAVE_TDM
    if (tid < 32)
      tdm_load_tile_bf16(kbf + kb + (long long)w0 * HD, sK, 128, 64, HD);
#else
    {
      const uint4* gk = reinterpret_cast<const uint4*>(
          kbf + kb + (long long)(w0 + ldRow) * HD + ldPart * 32);
      uint4* lk = reinterpret_cast<uint4*>(sK + ldRow * 128 + ldPart * 32);
      lk[0] = gk[0]; lk[1] = gk[1]; lk[2] = gk[2]; lk[3] = gk[3];
    }
#endif
    {
      uint4 vr[4];
      const uint4* gv = reinterpret_cast<const uint4*>(
          vbf + kb + (long long)(w0 + ldRow) * HD + ldPart * 32);
      vr[0] = gv[0]; vr[1] = gv[1]; vr[2] = gv[2]; vr[3] = gv[3];
      const u16* vs = reinterpret_cast<const u16*>(vr);
#pragma unroll
      for (int e = 0; e < 32; ++e)
        sVT[(ldPart * 32 + e) * 64 + ldRow] = vs[e];
    }
#ifdef HAVE_TDM
    if (tid < 32) __builtin_amdgcn_s_wait_tensorcnt(0);
#endif
    __syncthreads();

    // S = Q K^T  (16 x 64)
    v8f sacc[4];
#pragma unroll
    for (int nf = 0; nf < 4; ++nf) sacc[nf] = zero;
#pragma unroll
    for (int nf = 0; nf < 4; ++nf) {
      const u16* kp = sK + (nf * 16 + L15) * 128;
#pragma unroll
      for (int kk = 0; kk < 4; ++kk) {
        FragA fb;
        fb.q[0] = *reinterpret_cast<const uint4*>(kp + kk * 32 + hi * 8);
        fb.q[1] = *reinterpret_cast<const uint4*>(kp + kk * 32 + 16 + hi * 8);
        sacc[nf] = __builtin_amdgcn_wmma_f32_16x16x32_bf16(
            false, aq[kk].v, false, fb.v, (short)0, sacc[nf], false, false);
      }
    }

    // scale + ALiBi bias (depends only on cache-slot column)
#pragma unroll
    for (int nf = 0; nf < 4; ++nf) {
      float bias = ((float)(w0 + nf * 16 + L15) - shift) * slope;
#pragma unroll
      for (int j = 0; j < 8; ++j) sacc[nf][j] = sacc[nf][j] * ascale + bias;
    }

    // online softmax: rows at (vgpr j, half hi); cols across 16 lanes
#pragma unroll
    for (int j = 0; j < 8; ++j) {
      float mx = sacc[0][j];
#pragma unroll
      for (int nf = 1; nf < 4; ++nf) mx = fmaxf(mx, sacc[nf][j]);
#pragma unroll
      for (int off = 1; off < 16; off <<= 1)
        mx = fmaxf(mx, __shfl_xor(mx, off, 32));
      float mnew = fmaxf(mrow[j], mx);
      alpha[j] = __expf(mrow[j] - mnew);
      mrow[j] = mnew;
    }
#pragma unroll
    for (int j = 0; j < 8; ++j) {
      float sm = 0.f;
#pragma unroll
      for (int nf = 0; nf < 4; ++nf) {
        float p = __expf(sacc[nf][j] - mrow[j]);
        sacc[nf][j] = p;
        sm += p;
      }
#pragma unroll
      for (int off = 1; off < 16; off <<= 1) sm += __shfl_xor(sm, off, 32);
      lrow[j] = lrow[j] * alpha[j] + sm;
    }
#pragma unroll
    for (int f = 0; f < 8; ++f)
#pragma unroll
      for (int j = 0; j < 8; ++j) oacc[f][j] *= alpha[j];

    // transpose P (C-layout) -> A-layout via per-wave LDS region
    u16* pw = sP + wave * 16 * 64;
#pragma unroll
    for (int nf = 0; nf < 4; ++nf)
#pragma unroll
      for (int j = 0; j < 8; ++j)
        pw[(j + hi * 8) * 64 + nf * 16 + L15] = f2bf(sacc[nf][j]);
    __syncthreads();

    // O += P V  (2 k-steps x 8 hd-frags)
#pragma unroll
    for (int kf = 0; kf < 2; ++kf) {
      FragA pa;
      const u16* pp = pw + L15 * 64 + kf * 32;
      pa.q[0] = *reinterpret_cast<const uint4*>(pp + hi * 8);
      pa.q[1] = *reinterpret_cast<const uint4*>(pp + 16 + hi * 8);
#pragma unroll
      for (int nf = 0; nf < 8; ++nf) {
        FragA fv;
        const u16* vp = sVT + (nf * 16 + L15) * 64 + kf * 32;
        fv.q[0] = *reinterpret_cast<const uint4*>(vp + hi * 8);
        fv.q[1] = *reinterpret_cast<const uint4*>(vp + 16 + hi * 8);
        oacc[nf] = __builtin_amdgcn_wmma_f32_16x16x32_bf16(
            false, pa.v, false, fv.v, (short)0, oacc[nf], false, false);
      }
    }
    __syncthreads();
  }

  // epilogue: normalize and store [B,T,D] f32
  const int b = bh >> 4;
#pragma unroll
  for (int j = 0; j < 8; ++j) {
    float inv = 1.f / lrow[j];
    int t = m0 + j + hi * 8;
    float* op = outf + ((long long)(b * T + t)) * D + h * HD;
#pragma unroll
    for (int f = 0; f < 8; ++f) op[f * 16 + L15] = oacc[f][j] * inv;
  }
}

// ---------------------------------------------------------------------------
extern "C" void kernel_launch(void* const* d_in, const int* in_sizes, int n_in,
                              void* d_out, int out_size, void* d_ws, size_t ws_size,
                              hipStream_t stream) {
  (void)in_sizes; (void)n_in; (void)out_size; (void)ws_size;

  const float* x       = (const float*)d_in[0];
  const float* cache_k = (const float*)d_in[1];
  const float* cache_v = (const float*)d_in[2];
  const float* qkv_w   = (const float*)d_in[3];
  const float* qkv_b   = (const float*)d_in[4];
  const float* proj_w  = (const float*)d_in[5];
  const float* proj_b  = (const float*)d_in[6];
  const int*   pos     = (const int*)d_in[7];

  float* out  = (float*)d_out;          // [B,T,D]
  float* kout = out + NX;               // new_k [B,H,W,hd]
  float* vout = out + 2 * NX;           // new_v

  char* ws = (char*)d_ws;
  size_t off = 0;
  unsigned* scales = (unsigned*)(ws + off); off += 256;
  u16* xq    = (u16*)(ws + off); off += (size_t)NX * 2;
  u16* wqkv  = (u16*)(ws + off); off += (size_t)3 * D * D * 2;
  u16* wproj = (u16*)(ws + off); off += (size_t)D * D * 2;
  u16* qbf   = (u16*)(ws + off); off += (size_t)NKV * 2;
  u16* kbf   = (u16*)(ws + off); off += (size_t)NKV * 2;
  u16* vbf   = (u16*)(ws + off); off += (size_t)NKV * 2;
  float* attnf = (float*)(ws + off); off += (size_t)NX * 4;
  u16* obf   = (u16*)(ws + off); off += (size_t)NX * 2;

  // seed cache outputs with old cache (GEMM epilogue overwrites the T slots)
  (void)hipMemcpyAsync(kout, cache_k, (size_t)NKV * 4, hipMemcpyDeviceToDevice, stream);
  (void)hipMemcpyAsync(vout, cache_v, (size_t)NKV * 4, hipMemcpyDeviceToDevice, stream);

  k_init<<<1, 32, 0, stream>>>(scales);

  // fake-quant x -> bf16
  k_absmax<<<1024, 256, 0, stream>>>(x, NX, scales + 0);
  k_quant_bf16<<<4096, 256, 0, stream>>>(x, scales + 0, xq, NX);

  // weights + old cache -> bf16 shadows
  k_cvt_bf16<<<4096, 256, 0, stream>>>(qkv_w, wqkv, (long long)3 * D * D);
  k_cvt_bf16<<<4096, 256, 0, stream>>>(proj_w, wproj, (long long)D * D);
  k_cvt_bf16<<<4096, 256, 0, stream>>>(cache_k, kbf, NKV);
  k_cvt_bf16<<<4096, 256, 0, stream>>>(cache_v, vbf, NKV);

  // QKV GEMM: [4096 x 2048] x [2048 x 6144] + bias, fused scatter epilogue
  k_gemm_bf16<<<dim3(3 * D / 128, Mrows / 128), 256, 0, stream>>>(
      xq, wqkv, qkv_b, 3 * D, D, /*mode=*/1,
      nullptr, qbf, kout, vout, kbf, vbf);

  // attention
  k_attn<<<dim3(T / 128, Bb * Hh), 256, 0, stream>>>(qbf, kbf, vbf, pos, attnf);

  // fake-quant attention output -> bf16
  k_absmax<<<1024, 256, 0, stream>>>(attnf, NX, scales + 1);
  k_quant_bf16<<<4096, 256, 0, stream>>>(attnf, scales + 1, obf, NX);

  // output projection: [4096 x 2048] x [2048 x 2048] + bias -> d_out
  k_gemm_bf16<<<dim3(D / 128, Mrows / 128), 256, 0, stream>>>(
      obf, wproj, proj_b, D, D, /*mode=*/0,
      out, nullptr, nullptr, nullptr, nullptr, nullptr);
}
